// DeepSeekMoE_75771813036401
// MI455X (gfx1250) — matrix operand order repaired and checked by reading, not compiled, not run on hardware
//
#include <hip/hip_runtime.h>
#include <hip/hip_bf16.h>
#include <stdint.h>

typedef __bf16 bf16;
typedef bf16  v16bf __attribute__((ext_vector_type(16)));
typedef bf16  v8bf  __attribute__((ext_vector_type(8)));
typedef bf16  v2bf  __attribute__((ext_vector_type(2)));
typedef float v8f   __attribute__((ext_vector_type(8)));

#define DM 2048
#define DF 1408
#define TT 8192
#define NSH 2
#define NRT 16
#define KR 4

// tile geometry: BM=128, BN=128, BK=64; 256 threads = 8 wave32 waves (4 M x 2 N)
#define LDT 72                         // padded leading dim (bf16 elems)
#define ABUF_B (128 * LDT * 2)         // bytes per tile buffer
#define SMEM_B (4 * 128 * LDT * 2)     // 2xA + 2xB = 73728 B

// ---------------------------------------------------------------- helpers ---

__device__ __forceinline__ v8f wmma_bf16(v16bf a, v16bf b, v8f c) {
  return __builtin_amdgcn_wmma_f32_16x16x32_bf16(false, a, false, b, (short)0, c,
                                                 false, false);
}

// A-fragment 16x32 bf16, LDS row-major ld=LDT (ISA 7.12.2: lanes<16 hold
// K=0..7,16..23; lanes>=16 hold K=8..15,24..31 for row M=lane&15)
__device__ __forceinline__ v16bf frag_a(const bf16* base, int lane) {
  int row = lane & 15;
  int k0  = (lane < 16) ? 0 : 8;
  const bf16* p = base + row * LDT + k0;
  v8bf lo = *(const v8bf*)p;        // K = k0 .. k0+7
  v8bf hi = *(const v8bf*)(p + 16); // K = k0+16 .. k0+23
  return __builtin_shufflevector(lo, hi, 0,1,2,3,4,5,6,7,8,9,10,11,12,13,14,15);
}

// B-fragment 32x16 bf16, LDS column-major ld=LDT (lanes<16: K=0..7,16..23 of col
// N=lane; lanes>=16: K=8..15,24..31) -> two 8-bf16 runs per lane
__device__ __forceinline__ v16bf frag_b(const bf16* base, int lane) {
  int col = lane & 15;
  int k0  = (lane < 16) ? 0 : 8;
  const bf16* p = base + col * LDT + k0;
  v8bf lo = *(const v8bf*)p;
  v8bf hi = *(const v8bf*)(p + 16);
  return __builtin_shufflevector(lo, hi, 0,1,2,3,4,5,6,7,8,9,10,11,12,13,14,15);
}

__device__ __forceinline__ void async_lds_b128(uint32_t lds_off, const void* g) {
  asm volatile("global_load_async_to_lds_b128 %0, %1, off"
               :: "v"(lds_off), "v"((uint64_t)(uintptr_t)g)
               : "memory");
}
__device__ __forceinline__ void wait_async0() {
  asm volatile("s_wait_asynccnt 0x0" ::: "memory");
}

// ------------------------------------------------------------ GEMM kernel ---
// MODE 0: Hbf = relu(A[T,K]bf16 * W[K,N]f32 + b)  -> bf16
// MODE 1: Out[T,N]f32 += g_row * (A*W + b)
template <int MODE>
__global__ __launch_bounds__(256)
void ffn_gemm(const bf16* __restrict__ A, const float* __restrict__ W,
              const float* __restrict__ bias, void* __restrict__ Outv,
              const float* __restrict__ gates, int N, int K) {
  const int tid    = threadIdx.x;
  const int lane   = tid & 31;
  const int wid    = tid >> 5;
  const int wave_m = wid & 3;   // 4 waves along M (32 rows each)
  const int wave_n = wid >> 2;  // 2 waves along N (64 cols each)
  const int rowBase = blockIdx.y * 128;
  const int colBase = blockIdx.x * 128;

  __shared__ __align__(16) char smem[SMEM_B];  // mainloop tiles; epilogue reuses
  __shared__ float s_g[128];
  __shared__ int   s_any;

  // gating: early-out whole M-tile if all 128 gates are zero (top-k sparsity)
  if (gates != nullptr) {
    if (tid == 0) s_any = 0;
    __syncthreads();
    if (tid < 128) {
      float g = gates[(size_t)(rowBase + tid) * NRT];
      s_g[tid] = g;
      if (g > 0.0f) atomicOr(&s_any, 1);
    }
    __syncthreads();
    if (!s_any) return;
  } else {
    if (tid < 128) s_g[tid] = 1.0f;
  }

  // A tile: 128 rows x 64 bf16, async global->LDS, 16B chunks, 4 per thread
  auto issueA = [&](int buf, int kt) {
    uint32_t lbase = (uint32_t)(uintptr_t)(smem + buf * ABUF_B);
#pragma unroll
    for (int q = 0; q < 4; ++q) {
      int c = tid + q * 256;           // 0..1023
      int row = c >> 3, part = c & 7;  // 8 x 16B per row
      uint32_t loff = lbase + (uint32_t)(row * (LDT * 2) + part * 16);
      const void* gp = A + (size_t)(rowBase + row) * K + kt * 64 + part * 8;
      async_lds_b128(loff, gp);
    }
  };

  // W tile: 64(K) x 128(N) fp32 -> bf16, transposed col-major ld=LDT.
  // Each thread handles k/k+1 pairs of one float4 n-group -> packed b32 stores.
  float4 breg[8];
  auto loadB = [&](int kt) {
#pragma unroll
    for (int q = 0; q < 4; ++q) {
      int c  = tid + q * 256;          // 0..1023
      int k  = (c >> 5) * 2;           // even k row
      int n4 = c & 31;
      const float* p = W + (size_t)(kt * 64 + k) * N + colBase + n4 * 4;
      breg[2 * q]     = *(const float4*)p;
      breg[2 * q + 1] = *(const float4*)(p + N);
    }
  };
  auto storeB = [&](int buf) {
    bf16* Bb = (bf16*)(smem + (2 + buf) * ABUF_B);
#pragma unroll
    for (int q = 0; q < 4; ++q) {
      int c  = tid + q * 256;
      int k  = (c >> 5) * 2;
      int n  = (c & 31) * 4;
      const float* f0 = (const float*)&breg[2 * q];
      const float* f1 = (const float*)&breg[2 * q + 1];
#pragma unroll
      for (int i = 0; i < 4; ++i) {
        v2bf p2 = {(bf16)f0[i], (bf16)f1[i]};   // (k, k+1) adjacent in LDS
        *(v2bf*)(Bb + (n + i) * LDT + k) = p2;  // one ds_store_b32
      }
    }
  };

  const int NT = K >> 6;

  issueA(0, 0);
  loadB(0);
  storeB(0);
  wait_async0();
  __syncthreads();

  v8f acc[2][4];
#pragma unroll
  for (int i = 0; i < 2; ++i)
#pragma unroll
    for (int j = 0; j < 4; ++j) acc[i][j] = (v8f)0.0f;

  for (int kt = 0; kt < NT; ++kt) {
    const int cur = kt & 1, nxt = cur ^ 1;
    const bool more = (kt + 1 < NT);
    if (more) { issueA(nxt, kt + 1); loadB(kt + 1); }

    const bf16* Acur = (const bf16*)(smem + cur * ABUF_B);
    const bf16* Bcur = (const bf16*)(smem + (2 + cur) * ABUF_B);
#pragma unroll
    for (int ks = 0; ks < 2; ++ks) {   // two K=32 sub-steps -> 16 WMMA per sync
      const bf16* Ab = Acur + ks * 32;
      const bf16* Bb = Bcur + ks * 32;
      v16bf a0 = frag_a(Ab + (wave_m * 32 + 0) * LDT, lane);
      v16bf a1 = frag_a(Ab + (wave_m * 32 + 16) * LDT, lane);
#pragma unroll
      for (int j = 0; j < 4; ++j) {
        v16bf b = frag_b(Bb + (wave_n * 64 + j * 16) * LDT, lane);
        acc[0][j] = wmma_bf16(a0, b, acc[0][j]);
        acc[1][j] = wmma_bf16(a1, b, acc[1][j]);
      }
    }

    if (more) {
      storeB(nxt);
      wait_async0();
      __syncthreads();
    }
  }

  // ---- epilogue: scatter fragments to an LDS 128x128 tile, stream out 16B ----
  __syncthreads();   // all waves done reading mainloop LDS; safe to repurpose

  if (MODE == 0) {
    bf16* Ht = (bf16*)smem;            // 128x128 bf16 = 32 KB
#pragma unroll
    for (int i = 0; i < 2; ++i) {
      int m0 = wave_m * 32 + i * 16 + ((lane >= 16) ? 8 : 0);
#pragma unroll
      for (int j = 0; j < 4; ++j) {
        int n = wave_n * 64 + j * 16 + (lane & 15);
        float bc = bias[colBase + n];
#pragma unroll
        for (int r = 0; r < 8; ++r)
          Ht[(m0 + r) * 128 + n] = (bf16)fmaxf(acc[i][j][r] + bc, 0.0f);
      }
    }
    __syncthreads();
    bf16* Hg = (bf16*)Outv;
#pragma unroll
    for (int q = 0; q < 8; ++q) {      // 2048 x 16B chunks / 256 threads
      int c = tid + q * 256;
      int r = c >> 4, ch = c & 15;
      *(v8bf*)(Hg + (size_t)(rowBase + r) * N + colBase + ch * 8) =
          *(const v8bf*)(Ht + r * 128 + ch * 8);
    }
  } else {
    float* Ot = (float*)smem;          // 128x128 f32 = 64 KB (fits in SMEM_B)
#pragma unroll
    for (int i = 0; i < 2; ++i) {
      int m0 = wave_m * 32 + i * 16 + ((lane >= 16) ? 8 : 0);
#pragma unroll
      for (int j = 0; j < 4; ++j) {
        int n = wave_n * 64 + j * 16 + (lane & 15);
        float bc = bias[colBase + n];
#pragma unroll
        for (int r = 0; r < 8; ++r) {
          int m = m0 + r;
          Ot[m * 128 + n] = s_g[m] * (acc[i][j][r] + bc);
        }
      }
    }
    __syncthreads();
    float* Og = (float*)Outv;
#pragma unroll
    for (int q = 0; q < 16; ++q) {     // 4096 x 16B chunks / 256 threads
      int c = tid + q * 256;
      int r = c >> 5, ch = c & 31;
      float4* gp = (float4*)(Og + (size_t)(rowBase + r) * N + colBase + ch * 4);
      float4 o = *gp;
      float4 v = *(const float4*)(Ot + r * 128 + ch * 4);
      o.x += v.x; o.y += v.y; o.z += v.z; o.w += v.w;
      *gp = o;
    }
  }
}

// ------------------------------------------------------------ aux kernels ---

__global__ void prep_kernel(const float* __restrict__ u, float* __restrict__ out,
                            bf16* __restrict__ ubf) {
  size_t i = (size_t)blockIdx.x * 256 + threadIdx.x;
  float v = u[i];
  out[i] = v;           // residual init
  ubf[i] = (bf16)v;
}

__global__ void router_kernel(const float* __restrict__ u,
                              const float* __restrict__ cent,
                              const float* __restrict__ bias,
                              float* __restrict__ scores) {
  __shared__ float row[DM];
  int t = blockIdx.x;
  const float* ur = u + (size_t)t * DM;
  for (int i = threadIdx.x; i < DM; i += 256) row[i] = ur[i];
  __syncthreads();
  int wid = threadIdx.x >> 5, lane = threadIdx.x & 31;
  for (int e = wid; e < NRT; e += 8) {
    const float* c = cent + (size_t)e * DM;
    float s = 0.0f;
    for (int i = lane; i < DM; i += 32) s += row[i] * c[i];
#pragma unroll
    for (int m = 16; m > 0; m >>= 1) s += __shfl_xor(s, m, 32);
    if (lane == 0) scores[(size_t)t * NRT + e] = 1.0f / (1.0f + __expf(-s)) + bias[e];
  }
}

__global__ void topk_kernel(const float* __restrict__ scores,
                            float* __restrict__ gates) {
  int t = blockIdx.x * 256 + threadIdx.x;
  if (t >= TT) return;
  float v[NRT];
  bool used[NRT];
#pragma unroll
  for (int i = 0; i < NRT; ++i) { v[i] = scores[(size_t)t * NRT + i]; used[i] = false; }
  float g[NRT] = {};
  float sum = 0.0f;
#pragma unroll
  for (int k = 0; k < KR; ++k) {
    int am = -1; float mv = -1e30f;
#pragma unroll
    for (int i = 0; i < NRT; ++i)
      if (!used[i] && v[i] > mv) { mv = v[i]; am = i; }
    used[am] = true; g[am] = mv; sum += mv;
  }
  sum = fmaxf(sum, 1e-8f);
#pragma unroll
  for (int i = 0; i < NRT; ++i) gates[(size_t)t * NRT + i] = g[i] / sum;
}

// ---------------------------------------------------------------- launch ---

extern "C" void kernel_launch(void* const* d_in, const int* in_sizes, int n_in,
                              void* d_out, int out_size, void* d_ws, size_t ws_size,
                              hipStream_t stream) {
  const float* u    = (const float*)d_in[0];
  const float* sw1  = (const float*)d_in[1];
  const float* sb1  = (const float*)d_in[2];
  const float* sw2  = (const float*)d_in[3];
  const float* sb2  = (const float*)d_in[4];
  const float* rw1  = (const float*)d_in[5];
  const float* rb1  = (const float*)d_in[6];
  const float* rw2  = (const float*)d_in[7];
  const float* rb2  = (const float*)d_in[8];
  const float* cent = (const float*)d_in[9];
  const float* rbias= (const float*)d_in[10];
  float* out = (float*)d_out;

  char* ws = (char*)d_ws;
  bf16* ubf     = (bf16*)(ws);                                   // T*DM bf16 = 32 MB
  bf16* hbf     = (bf16*)(ws + (size_t)TT * DM * 2);             // T*DF bf16 = 22 MB
  float* scores = (float*)(ws + (size_t)TT * DM * 2 + (size_t)TT * DF * 2);
  float* gates  = scores + (size_t)TT * NRT;

  prep_kernel<<<(TT * DM) / 256, 256, 0, stream>>>(u, out, ubf);
  router_kernel<<<TT, 256, 0, stream>>>(u, cent, rbias, scores);
  topk_kernel<<<TT / 256, 256, 0, stream>>>(scores, gates);

  dim3 blk(256);
  dim3 g1(DF / 128, TT / 128);   // ffn layer 1: N = D_FF
  dim3 g2(DM / 128, TT / 128);   // ffn layer 2: N = D_MODEL

  for (int e = 0; e < NSH; ++e) {
    ffn_gemm<0><<<g1, blk, 0, stream>>>(ubf, sw1 + (size_t)e * DM * DF,
                                        sb1 + (size_t)e * DF, (void*)hbf,
                                        nullptr, DF, DM);
    ffn_gemm<1><<<g2, blk, 0, stream>>>(hbf, sw2 + (size_t)e * DF * DM,
                                        sb2 + (size_t)e * DM, (void*)out,
                                        nullptr, DM, DF);
  }
  for (int e = 0; e < NRT; ++e) {
    ffn_gemm<0><<<g1, blk, 0, stream>>>(ubf, rw1 + (size_t)e * DM * DF,
                                        rb1 + (size_t)e * DF, (void*)hbf,
                                        gates + e, DF, DM);
    ffn_gemm<1><<<g2, blk, 0, stream>>>(hbf, rw2 + (size_t)e * DF * DM,
                                        rb2 + (size_t)e * DM, (void*)out,
                                        gates + e, DM, DF);
  }
}
